// MPNN_69664369541682
// MI455X (gfx1250) — compile-verified, matrix operand order checked by reading, not run on hardware
//
#include <hip/hip_runtime.h>

#define N_ 30000
#define E_ 60000
#define B_ 500

typedef __attribute__((ext_vector_type(16))) __bf16 bf16x16;
typedef __attribute__((ext_vector_type(8)))  float  f32x8;

// ---------- helpers ----------
__device__ inline __bf16 f2bf(float f) { return (__bf16)f; }   // native v_cvt_pk_bf16_f32

__device__ inline f32x8 zero8() {
  f32x8 z;
  #pragma unroll
  for (int i = 0; i < 8; ++i) z[i] = 0.f;
  return z;
}

__device__ inline bf16x16 splat16(__bf16 s) {
  bf16x16 v;
  #pragma unroll
  for (int i = 0; i < 16; ++i) v[i] = s;
  return v;
}

__device__ inline f32x8 wmma_bf(bf16x16 a, bf16x16 b, f32x8 c) {
  return __builtin_amdgcn_wmma_f32_16x16x32_bf16(false, a, false, b, (short)0, c, false, false);
}

// A fragment (16x32 bf16) from row-major f32 (global or LDS), K base kb.
// lane L: row m = L&15, half = L>>4. j<8 -> k = kb+8*half+j ; j>=8 -> k = kb+16+8*half+(j-8)
__device__ inline bf16x16 load_a_f32(const float* src, int ld, int kb, int lane) {
  int half = lane >> 4, m = lane & 15;
  const float* p = src + m * ld + kb + 8 * half;
  bf16x16 a;
  #pragma unroll
  for (int j = 0; j < 8; ++j) a[j] = f2bf(p[j]);
  #pragma unroll
  for (int j = 0; j < 8; ++j) a[j + 8] = f2bf(p[16 + j]);
  return a;
}

// B fragment from fragment-ordered bf16 storage: frag f, 32 lanes x 16 elems.
__device__ inline bf16x16 fragld(const __bf16* base, int f, int lane) {
  return ((const bf16x16*)base)[f * 32 + lane];
}

__device__ inline float sigmoidf_(float x) { return 1.f / (1.f + __expf(-x)); }

// ---------- weight pre-swizzle into fragment order ----------
// frag element (f = t*2+kc, lane, j): B[k][n] with k = kc*32 + kmap(j,half), n = t*16 + (lane&15)
__global__ void prep_frags(const float* W_edge, const float* b_edge,
                           const float* root, const float* Wih, const float* Whh,
                           const float* Wp,
                           __bf16* fEdge, __bf16* fRoot, __bf16* fIh, __bf16* fHh,
                           __bf16* fProj) {
  int tid = blockIdx.x * blockDim.x + threadIdx.x;
  int stride = gridDim.x * blockDim.x;
  // edge: 17 k-slabs x 8 frags (k=16 is b_edge reshaped 64x64)
  for (int i = tid; i < 17 * 8 * 512; i += stride) {
    int j = i & 15, lane = (i >> 4) & 31, f = (i >> 9) & 7, k = i >> 12;
    int kc = f & 1, t = f >> 1, half = lane >> 4, nn = lane & 15;
    int hh = kc * 32 + (j < 8 ? 8 * half + j : 16 + 8 * half + (j - 8));
    int col = t * 16 + nn;
    float v = (k < 16) ? W_edge[k * 4096 + hh * 64 + col] : b_edge[hh * 64 + col];
    fEdge[i] = f2bf(v);
  }
  // root (row-major [64][64]) : 8 frags
  for (int i = tid; i < 8 * 512; i += stride) {
    int j = i & 15, lane = (i >> 4) & 31, f = i >> 9;
    int kc = f & 1, t = f >> 1, half = lane >> 4, nn = lane & 15;
    int k = kc * 32 + (j < 8 ? 8 * half + j : 16 + 8 * half + (j - 8));
    fRoot[i] = f2bf(root[k * 64 + t * 16 + nn]);
  }
  // gru W_ih / W_hh ([192][64], used transposed): 24 frags each
  for (int i = tid; i < 24 * 512; i += stride) {
    int j = i & 15, lane = (i >> 4) & 31, f = i >> 9;
    int kc = f & 1, t = f >> 1, half = lane >> 4, nn = lane & 15;
    int k = kc * 32 + (j < 8 ? 8 * half + j : 16 + 8 * half + (j - 8));
    fIh[i] = f2bf(Wih[(t * 16 + nn) * 64 + k]);
    fHh[i] = f2bf(Whh[(t * 16 + nn) * 64 + k]);
  }
  // W_proj (row-major [64][64]) : 8 frags
  for (int i = tid; i < 8 * 512; i += stride) {
    int j = i & 15, lane = (i >> 4) & 31, f = i >> 9;
    int kc = f & 1, t = f >> 1, half = lane >> 4, nn = lane & 15;
    int k = kc * 32 + (j < 8 ? 8 * half + j : 16 + 8 * half + (j - 8));
    fProj[i] = f2bf(Wp[k * 64 + t * 16 + nn]);
  }
}

// ---------- misc small kernels ----------
__global__ void zero_f32(float* p, int n) {
  for (int i = blockIdx.x * blockDim.x + threadIdx.x; i < n; i += gridDim.x * blockDim.x)
    p[i] = 0.f;
}
__global__ void ranges_init(int* gs, int* ge) {
  int b = blockIdx.x * blockDim.x + threadIdx.x;
  if (b < B_) { gs[b] = N_; ge[b] = 0; }
}
__global__ void ranges_kernel(const int* batch, int* gs, int* ge) {
  int i = blockIdx.x * blockDim.x + threadIdx.x;
  if (i < N_) { int b = batch[i]; atomicMin(&gs[b], i); atomicMax(&ge[b], i + 1); }
}

// ---------- h0 = relu(x @ W_proj + b_proj) ----------
__global__ __launch_bounds__(256) void proj_kernel(const float* x, const __bf16* fProj,
                                                   const float* bp, float* h0, float* h) {
  int lane = threadIdx.x & 31, wave = threadIdx.x >> 5, wpb = blockDim.x >> 5;
  int half = lane >> 4, nn = lane & 15;
  // per-lane bias values (col = t*16+nn), hoisted out of the tile loop
  float bpv[4];
  #pragma unroll
  for (int t = 0; t < 4; ++t) bpv[t] = bp[t * 16 + nn];
  for (int tile = blockIdx.x * wpb + wave; tile < N_ / 16; tile += gridDim.x * wpb) {
    int m0 = tile * 16;
    bf16x16 a0 = load_a_f32(x + (size_t)m0 * 64, 64, 0, lane);
    bf16x16 a1 = load_a_f32(x + (size_t)m0 * 64, 64, 32, lane);
    f32x8 acc[4];
    #pragma unroll
    for (int t = 0; t < 4; ++t) {
      acc[t] = zero8();
      acc[t] = wmma_bf(a0, fragld(fProj, t * 2 + 0, lane), acc[t]);
      acc[t] = wmma_bf(a1, fragld(fProj, t * 2 + 1, lane), acc[t]);
    }
    #pragma unroll
    for (int t = 0; t < 4; ++t) {
      #pragma unroll
      for (int r = 0; r < 8; ++r) {
        int row = m0 + r + 8 * half, col = t * 16 + nn;
        float v = fmaxf(acc[t][r] + bpv[t], 0.f);
        h0[(size_t)row * 64 + col] = v;
        h[(size_t)row * 64 + col] = v;
      }
    }
  }
}

// ---------- fused edge-network + message + scatter-add ----------
// msg = sum_k diag(ea[:,k]) * (N_src @ W_edge_k)  +  N_src @ reshape(b_edge)
__global__ __launch_bounds__(256) void msg_kernel(const int* eidx, const float* ea,
                                                  const __bf16* fEdge, const float* h,
                                                  float* aggr) {
  int lane = threadIdx.x & 31, wave = threadIdx.x >> 5, wpb = blockDim.x >> 5;
  int half = lane >> 4, mr = lane & 15, nn = lane & 15;
  for (int tile = blockIdx.x * wpb + wave; tile < E_ / 16; tile += gridDim.x * wpb) {
    int e = tile * 16 + mr;
    int src = eidx[e];
    const float* nrow = h + (size_t)src * 64;
    // node row in A-fragment order, converted to bf16 once
    bf16x16 nr0, nr1;
    #pragma unroll
    for (int j = 0; j < 8; ++j) {
      nr0[j]     = f2bf(nrow[8 * half + j]);
      nr0[8 + j] = f2bf(nrow[16 + 8 * half + j]);
      nr1[j]     = f2bf(nrow[32 + 8 * half + j]);
      nr1[8 + j] = f2bf(nrow[48 + 8 * half + j]);
    }
    __bf16 eab[17];
    #pragma unroll
    for (int k = 0; k < 16; ++k) eab[k] = f2bf(ea[(size_t)e * 16 + k]);
    eab[16] = (__bf16)1.0f;                  // b_edge virtual channel
    f32x8 acc[4];
    #pragma unroll
    for (int t = 0; t < 4; ++t) acc[t] = zero8();
    for (int k = 0; k < 17; ++k) {
      bf16x16 sv = splat16(eab[k]);
      bf16x16 A0 = nr0 * sv;                 // packed bf16 multiplies
      bf16x16 A1 = nr1 * sv;
      const __bf16* base = fEdge + (size_t)k * 8 * 512;
      #pragma unroll
      for (int t = 0; t < 4; ++t) {
        acc[t] = wmma_bf(A0, fragld(base, t * 2 + 0, lane), acc[t]);
        acc[t] = wmma_bf(A1, fragld(base, t * 2 + 1, lane), acc[t]);
      }
    }
    int dsts[8];
    #pragma unroll
    for (int r = 0; r < 8; ++r) dsts[r] = eidx[E_ + tile * 16 + r + 8 * half];
    #pragma unroll
    for (int r = 0; r < 8; ++r) {
      #pragma unroll
      for (int t = 0; t < 4; ++t)
        atomicAdd(&aggr[(size_t)dsts[r] * 64 + t * 16 + nn], acc[t][r]);
    }
  }
}

// ---------- m = relu(aggr + h@root + bias); GRU(h <- m, h) ----------
__global__ __launch_bounds__(256) void update_kernel(float* h, const float* aggr,
                                                     const __bf16* fRoot, const float* cb,
                                                     const __bf16* fIh, const __bf16* fHh,
                                                     const float* bih, const float* bhh) {
  __shared__ float sM[8 * 16 * 64];          // per-wave 16x64 transpose staging
  int lane = threadIdx.x & 31, wave = threadIdx.x >> 5, wpb = blockDim.x >> 5;
  int half = lane >> 4, nn = lane & 15;
  float* myM = sM + wave * 16 * 64;
  // per-lane biases (hc = t*16+nn), hoisted out of the tile loop
  float cbv[4], brz[8], bnI[4], bnH[4];
  #pragma unroll
  for (int t = 0; t < 4; ++t) {
    int hc = t * 16 + nn;
    cbv[t]     = cb[hc];
    brz[t]     = bih[hc]      + bhh[hc];        // r-gate bias
    brz[4 + t] = bih[64 + hc] + bhh[64 + hc];   // z-gate bias
    bnI[t]     = bih[128 + hc];
    bnH[t]     = bhh[128 + hc];
  }
  for (int tile = blockIdx.x * wpb + wave; tile < N_ / 16; tile += gridDim.x * wpb) {
    int m0 = tile * 16;
    bf16x16 a0 = load_a_f32(h + (size_t)m0 * 64, 64, 0, lane);   // hidden A frags
    bf16x16 a1 = load_a_f32(h + (size_t)m0 * 64, 64, 32, lane);
    // preload aggr + old hidden tile values (batched loads, one wait)
    float agv[4][8], hold[4][8];
    #pragma unroll
    for (int t = 0; t < 4; ++t) {
      #pragma unroll
      for (int r = 0; r < 8; ++r) {
        size_t idx = (size_t)(m0 + r + 8 * half) * 64 + t * 16 + nn;
        agv[t][r]  = aggr[idx];
        hold[t][r] = h[idx];
      }
    }
    // node @ root
    f32x8 accR[4];
    #pragma unroll
    for (int t = 0; t < 4; ++t) {
      accR[t] = zero8();
      accR[t] = wmma_bf(a0, fragld(fRoot, t * 2 + 0, lane), accR[t]);
      accR[t] = wmma_bf(a1, fragld(fRoot, t * 2 + 1, lane), accR[t]);
    }
    // m = relu(aggr + node@root + conv_bias), stage to LDS for A-layout reuse
    #pragma unroll
    for (int t = 0; t < 4; ++t) {
      #pragma unroll
      for (int r = 0; r < 8; ++r) {
        int row = r + 8 * half, col = t * 16 + nn;
        myM[row * 64 + col] = fmaxf(accR[t][r] + agv[t][r] + cbv[t], 0.f);
      }
    }
    bf16x16 ma0 = load_a_f32(myM, 64, 0, lane);
    bf16x16 ma1 = load_a_f32(myM, 64, 32, lane);
    // gi+gh for r,z gates (cols 0..127); gi, gh separate for n gate (cols 128..191)
    f32x8 rz[8], gin[4], ghn[4];
    #pragma unroll
    for (int t = 0; t < 8; ++t) {
      f32x8 z = zero8();
      z = wmma_bf(ma0, fragld(fIh, t * 2 + 0, lane), z);
      z = wmma_bf(ma1, fragld(fIh, t * 2 + 1, lane), z);
      z = wmma_bf(a0,  fragld(fHh, t * 2 + 0, lane), z);
      z = wmma_bf(a1,  fragld(fHh, t * 2 + 1, lane), z);
      rz[t] = z;
    }
    #pragma unroll
    for (int t = 0; t < 4; ++t) {
      f32x8 z = zero8();
      z = wmma_bf(ma0, fragld(fIh, (8 + t) * 2 + 0, lane), z);
      z = wmma_bf(ma1, fragld(fIh, (8 + t) * 2 + 1, lane), z);
      gin[t] = z;
      f32x8 w = zero8();
      w = wmma_bf(a0, fragld(fHh, (8 + t) * 2 + 0, lane), w);
      w = wmma_bf(a1, fragld(fHh, (8 + t) * 2 + 1, lane), w);
      ghn[t] = w;
    }
    // gate math + in-place hidden update (all operands already in registers)
    #pragma unroll
    for (int t = 0; t < 4; ++t) {
      #pragma unroll
      for (int r = 0; r < 8; ++r) {
        int row = m0 + r + 8 * half, hc = t * 16 + nn;
        float rg = sigmoidf_(rz[t][r]     + brz[t]);
        float zg = sigmoidf_(rz[t + 4][r] + brz[4 + t]);
        float ng = tanhf((gin[t][r] + bnI[t]) + rg * (ghn[t][r] + bnH[t]));
        h[(size_t)row * 64 + hc] = (1.f - zg) * ng + zg * hold[t][r];
      }
    }
  }
}

// ---------- node classifier head ----------
__global__ void cls_kernel(const float* h, const float* Wc, const float* bc, float* out) {
  int i = blockIdx.x * blockDim.x + threadIdx.x;
  if (i >= N_) return;
  const float* r = h + (size_t)i * 64;
  float acc = bc[0];
  #pragma unroll
  for (int d = 0; d < 64; ++d) acc += r[d] * Wc[d];
  out[i] = acc;
}

// ---------- Set2Set ----------
__global__ void lstm_kernel(float* qstar, float* hs, float* cs,
                            const float* Wih, const float* Whh,
                            const float* bih, const float* bhh) {
  int b = blockIdx.x, d = threadIdx.x;          // d in 0..127
  __shared__ float sq[256], sh[128];
  for (int i = threadIdx.x; i < 256; i += 128) sq[i] = qstar[(size_t)b * 256 + i];
  sh[d] = hs[(size_t)b * 128 + d];
  __syncthreads();
  float g[4];
  #pragma unroll
  for (int gi = 0; gi < 4; ++gi) {
    int j = gi * 128 + d;
    float acc = bih[j] + bhh[j];
    for (int k = 0; k < 256; ++k) acc += sq[k] * Wih[(size_t)j * 256 + k];
    for (int k = 0; k < 128; ++k) acc += sh[k] * Whh[(size_t)j * 128 + k];
    g[gi] = acc;
  }
  float ig = sigmoidf_(g[0]), fg = sigmoidf_(g[1]);
  float gg = tanhf(g[2]),     og = sigmoidf_(g[3]);
  float c = fg * cs[(size_t)b * 128 + d] + ig * gg;
  float hn = og * tanhf(c);
  cs[(size_t)b * 128 + d] = c;
  hs[(size_t)b * 128 + d] = hn;
  qstar[(size_t)b * 256 + d] = hn;              // q half of q_star
}

__global__ void attn_e(const float* h0, const float* h, const float* hs,
                       const int* batch, float* e) {
  int i = blockIdx.x * blockDim.x + threadIdx.x;
  if (i >= N_) return;
  const float* q = hs + (size_t)batch[i] * 128;
  float acc = 0.f;
  #pragma unroll 4
  for (int d = 0; d < 64; ++d) acc += h0[(size_t)i * 64 + d] * q[d];
  #pragma unroll 4
  for (int d = 0; d < 64; ++d) acc += h[(size_t)i * 64 + d] * q[64 + d];
  e[i] = acc;
}

__global__ void attn_reduce(const float* e, const float* h0, const float* h,
                            const int* gs, const int* ge, float* qstar) {
  int b = blockIdx.x, tid = threadIdx.x;        // 128 threads
  int s = gs[b], t = ge[b];
  __shared__ float red[128];
  float mx = -3.0e38f;
  for (int i = s + tid; i < t; i += 128) mx = fmaxf(mx, e[i]);
  red[tid] = mx; __syncthreads();
  for (int o = 64; o > 0; o >>= 1) {
    if (tid < o) red[tid] = fmaxf(red[tid], red[tid + o]);
    __syncthreads();
  }
  float m = red[0]; __syncthreads();
  float sm = 0.f;
  for (int i = s + tid; i < t; i += 128) sm += __expf(e[i] - m);
  red[tid] = sm; __syncthreads();
  for (int o = 64; o > 0; o >>= 1) {
    if (tid < o) red[tid] += red[tid + o];
    __syncthreads();
  }
  float denom = red[0];
  float rv = 0.f;
  for (int i = s; i < t; ++i) {
    float w = __expf(e[i] - m);
    float xv = (tid < 64) ? h0[(size_t)i * 64 + tid] : h[(size_t)i * 64 + tid - 64];
    rv += w * xv;
  }
  qstar[(size_t)b * 256 + 128 + tid] = (t > s) ? (rv / denom) : 0.f;
}

// ---------- final head ----------
__global__ void sparsify(const float* qstar, const float* Wsp, const float* bsp,
                         const float* prelu, float* gf) {
  int idx = blockIdx.x * blockDim.x + threadIdx.x;
  if (idx >= B_ * 1024) return;
  int b = idx >> 10, j = idx & 1023;
  float acc = bsp[j];
  for (int k = 0; k < 256; ++k) acc += qstar[(size_t)b * 256 + k] * Wsp[(size_t)k * 1024 + j];
  float a = prelu[0];
  gf[idx] = (acc >= 0.f) ? acc : a * acc;
}
__global__ void y1_kernel(const float* gf, const float* W1, const float* b1, float* y1) {
  int idx = blockIdx.x * blockDim.x + threadIdx.x;
  if (idx >= B_ * 128) return;
  int b = idx >> 7, j = idx & 127;
  float acc = b1[j];
  for (int k = 0; k < 1024; ++k) acc += gf[(size_t)b * 1024 + k] * W1[(size_t)k * 128 + j];
  y1[idx] = fmaxf(acc, 0.f);
}
__global__ void y2_kernel(const float* y1, const float* W2, const float* b2, float* out) {
  int b = blockIdx.x * blockDim.x + threadIdx.x;
  if (b >= B_) return;
  float acc = b2[0];
  #pragma unroll 4
  for (int k = 0; k < 128; ++k) acc += y1[(size_t)b * 128 + k] * W2[k];
  out[b] = acc;
}

// ---------- host ----------
extern "C" void kernel_launch(void* const* d_in, const int* in_sizes, int n_in,
                              void* d_out, int out_size, void* d_ws, size_t ws_size,
                              hipStream_t stream) {
  (void)in_sizes; (void)n_in; (void)out_size; (void)ws_size;
  const float* x      = (const float*)d_in[0];
  const int*   eidx   = (const int*)d_in[1];
  const float* ea     = (const float*)d_in[2];
  const int*   batch  = (const int*)d_in[3];
  const float* W_proj = (const float*)d_in[4];  const float* b_proj = (const float*)d_in[5];
  const float* W_edge = (const float*)d_in[6];  const float* b_edge = (const float*)d_in[7];
  const float* root   = (const float*)d_in[8];  const float* cbias  = (const float*)d_in[9];
  const float* gWih   = (const float*)d_in[10]; const float* gWhh   = (const float*)d_in[11];
  const float* gbih   = (const float*)d_in[12]; const float* gbhh   = (const float*)d_in[13];
  const float* W_cls  = (const float*)d_in[14]; const float* b_cls  = (const float*)d_in[15];
  const float* lWih   = (const float*)d_in[16]; const float* lWhh   = (const float*)d_in[17];
  const float* lbih   = (const float*)d_in[18]; const float* lbhh   = (const float*)d_in[19];
  const float* W_sp   = (const float*)d_in[20]; const float* b_sp   = (const float*)d_in[21];
  const float* prelu  = (const float*)d_in[22];
  const float* W_y1   = (const float*)d_in[23]; const float* b_y1   = (const float*)d_in[24];
  const float* W_y2   = (const float*)d_in[25]; const float* b_y2   = (const float*)d_in[26];
  float* out = (float*)d_out;

  char* ws = (char*)d_ws; size_t off = 0;
  auto carve = [&](size_t bytes) -> void* {
    off = (off + 255) & ~size_t(255);
    void* p = ws + off; off += bytes; return p;
  };
  float* h0    = (float*)carve((size_t)N_ * 64 * 4);
  float* h     = (float*)carve((size_t)N_ * 64 * 4);
  float* aggr  = (float*)carve((size_t)N_ * 64 * 4);
  float* evec  = (float*)carve((size_t)N_ * 4);
  float* qstar = (float*)carve((size_t)B_ * 256 * 4);
  float* hs    = (float*)carve((size_t)B_ * 128 * 4);
  float* cs    = (float*)carve((size_t)B_ * 128 * 4);
  float* gfb   = (float*)carve((size_t)B_ * 1024 * 4);
  float* y1b   = (float*)carve((size_t)B_ * 128 * 4);
  int* gs = (int*)carve((size_t)B_ * 4);
  int* ge = (int*)carve((size_t)B_ * 4);
  __bf16* fEdge = (__bf16*)carve((size_t)17 * 8 * 512 * 2);
  __bf16* fRoot = (__bf16*)carve((size_t)8 * 512 * 2);
  __bf16* fIh   = (__bf16*)carve((size_t)24 * 512 * 2);
  __bf16* fHh   = (__bf16*)carve((size_t)24 * 512 * 2);
  __bf16* fProj = (__bf16*)carve((size_t)8 * 512 * 2);

  prep_frags<<<128, 256, 0, stream>>>(W_edge, b_edge, root, gWih, gWhh, W_proj,
                                      fEdge, fRoot, fIh, fHh, fProj);
  ranges_init<<<(B_ + 255) / 256, 256, 0, stream>>>(gs, ge);
  ranges_kernel<<<(N_ + 255) / 256, 256, 0, stream>>>(batch, gs, ge);
  proj_kernel<<<(N_ / 16 + 7) / 8, 256, 0, stream>>>(x, fProj, b_proj, h0, h);

  for (int s = 0; s < 3; ++s) {
    zero_f32<<<256, 256, 0, stream>>>(aggr, N_ * 64);
    msg_kernel<<<(E_ / 16 + 7) / 8, 256, 0, stream>>>(eidx, ea, fEdge, h, aggr);
    update_kernel<<<(N_ / 16 + 7) / 8, 256, 0, stream>>>(h, aggr, fRoot, cbias,
                                                         fIh, fHh, gbih, gbhh);
  }
  cls_kernel<<<(N_ + 255) / 256, 256, 0, stream>>>(h, W_cls, b_cls, out);

  zero_f32<<<64, 256, 0, stream>>>(qstar, B_ * 256);
  zero_f32<<<64, 256, 0, stream>>>(hs, B_ * 128);
  zero_f32<<<64, 256, 0, stream>>>(cs, B_ * 128);
  for (int it = 0; it < 3; ++it) {
    lstm_kernel<<<B_, 128, 0, stream>>>(qstar, hs, cs, lWih, lWhh, lbih, lbhh);
    attn_e<<<(N_ + 255) / 256, 256, 0, stream>>>(h0, h, hs, batch, evec);
    attn_reduce<<<B_, 128, 0, stream>>>(evec, h0, h, gs, ge, qstar);
  }
  sparsify<<<(B_ * 1024 + 255) / 256, 256, 0, stream>>>(qstar, W_sp, b_sp, prelu, gfb);
  y1_kernel<<<(B_ * 128 + 255) / 256, 256, 0, stream>>>(gfb, W_y1, b_y1, y1b);
  y2_kernel<<<(B_ + 255) / 256, 256, 0, stream>>>(y1b, W_y2, b_y2, out + N_);
}
